// SS2D_1709396984501
// MI455X (gfx1250) — compile-verified, matrix operand order checked by reading, not run on hardware
//
#include <hip/hip_runtime.h>
#include <hip/hip_bf16.h>

// ---------------- problem constants (from reference) ----------------
#define BB   8
#define HH   64
#define WW   64
#define LL   4096           // HH*WW
#define DM   192            // d_model
#define DI   384            // d_inner
#define KD4  4              // directions
#define NS   16             // d_state
#define DTR  12             // dt_rank
#define CPAD 48             // dt_rank + 2*d_state (=44) padded to 48
#define BL   (BB*LL)        // 32768 tokens
#define CS   128            // scan chunk length
#define NCH  (LL/CS)        // 32 chunks per (k,b) sequence

typedef __attribute__((ext_vector_type(16))) __bf16 v16bf;
typedef __attribute__((ext_vector_type(8)))  float  v8f;

__device__ __forceinline__ __bf16 f2bf(float x) {
  unsigned u = __builtin_bit_cast(unsigned, x);
  unsigned r = u + 0x7FFFu + ((u >> 16) & 1u);   // round-to-nearest-even
  unsigned short hs = (unsigned short)(r >> 16);
  return __builtin_bit_cast(__bf16, hs);
}

// ---------------- generic f32 -> bf16 ----------------
__global__ void k_f32_to_bf16(const float* __restrict__ s, __bf16* __restrict__ d, size_t n) {
  size_t i = (size_t)blockIdx.x * blockDim.x + threadIdx.x;
  if (i < n) d[i] = f2bf(s[i]);
}

// ---------------- x_proj weight: (4,44,384) -> bf16 (4,48,384), zero-pad rows 44..47 ----------------
__global__ void k_pad_xprojw(const float* __restrict__ w, __bf16* __restrict__ o) {
  int i = blockIdx.x * blockDim.x + threadIdx.x;
  if (i >= KD4 * CPAD * DI) return;
  int dcol = i % DI;
  int c    = (i / DI) % CPAD;
  int k    = i / (DI * CPAD);
  float v = (c < DTR + 2 * NS) ? w[((size_t)k * (DTR + 2 * NS) + c) * DI + dcol] : 0.f;
  o[i] = f2bf(v);
}

__global__ void k_zero(float* __restrict__ p, size_t n) {
  size_t i = (size_t)blockIdx.x * blockDim.x + threadIdx.x;
  if (i < n) p[i] = 0.f;
}

// ---------------- WMMA bf16 GEMM, 2x2 tiles/wave: C(M,N) = A(M,K) * B(N,K)^T + bias ----------------
// A-fragment layout per ISA 7.12.2 (16-bit A 16x32): lanes 0-15 -> rows, lane half
// selects K-base (0 or 8); vector elems 0..7 = K[k0..k0+7], 8..15 = K[k0+16..k0+23].
// B operand mirrors the mapping (lane = column), so loading W rows (n,k) identically
// supplies B = W^T. M must be a multiple of 32 (true for all calls: M = 32768).
// The N-edge case is hoisted OUT of the K loop so the full-tile path is a
// straight-line 4-WMMA clause with no per-iteration EXEC manipulation.
__device__ __forceinline__ v16bf load_frag16x32(const __bf16* __restrict__ base, int ld, int lane) {
  int r  = lane & 15;
  int k0 = (lane >> 4) << 3;
  const __bf16* p = base + (size_t)r * ld + k0;
  v16bf f;
#pragma unroll
  for (int i = 0; i < 8; i++) f[i] = p[i];
#pragma unroll
  for (int i = 0; i < 8; i++) f[8 + i] = p[16 + i];
  return f;
}

__global__ void k_gemm_wmma(const __bf16* __restrict__ A, const __bf16* __restrict__ Bm,
                            const float* __restrict__ bias, float* __restrict__ C,
                            int M, int N, int K) {
  int wid = (int)(((size_t)blockIdx.x * blockDim.x + threadIdx.x) >> 5);
  int ntiles = N >> 4;
  int n2t = (ntiles + 1) >> 1;
  int mt2 = wid / n2t;
  int nt2 = wid - mt2 * n2t;
  int mt = mt2 * 2, nt = nt2 * 2;
  if (mt * 16 >= M) return;
  const bool hasN1 = (nt + 1) < ntiles; // wave-uniform, loop-invariant
  int lane = threadIdx.x & 31;
  const __bf16* Ab0 = A  + (size_t)mt * 16 * K;
  const __bf16* Ab1 = Ab0 + (size_t)16 * K;
  const __bf16* Bb0 = Bm + (size_t)nt * 16 * K;
  const __bf16* Bb1 = Bb0 + (size_t)16 * K;
  v8f acc00 = {0.f,0.f,0.f,0.f,0.f,0.f,0.f,0.f};
  v8f acc01 = acc00, acc10 = acc00, acc11 = acc00;

  if (hasN1) {
    for (int kk = 0; kk < K; kk += 32) {
      v16bf a0 = load_frag16x32(Ab0 + kk, K, lane);
      v16bf a1 = load_frag16x32(Ab1 + kk, K, lane);
      v16bf b0 = load_frag16x32(Bb0 + kk, K, lane);
      v16bf b1 = load_frag16x32(Bb1 + kk, K, lane);
      acc00 = __builtin_amdgcn_wmma_f32_16x16x32_bf16(false, a0, false, b0, (short)0, acc00, false, false);
      acc10 = __builtin_amdgcn_wmma_f32_16x16x32_bf16(false, a1, false, b0, (short)0, acc10, false, false);
      acc01 = __builtin_amdgcn_wmma_f32_16x16x32_bf16(false, a0, false, b1, (short)0, acc01, false, false);
      acc11 = __builtin_amdgcn_wmma_f32_16x16x32_bf16(false, a1, false, b1, (short)0, acc11, false, false);
    }
  } else {
    for (int kk = 0; kk < K; kk += 32) {
      v16bf a0 = load_frag16x32(Ab0 + kk, K, lane);
      v16bf a1 = load_frag16x32(Ab1 + kk, K, lane);
      v16bf b0 = load_frag16x32(Bb0 + kk, K, lane);
      acc00 = __builtin_amdgcn_wmma_f32_16x16x32_bf16(false, a0, false, b0, (short)0, acc00, false, false);
      acc10 = __builtin_amdgcn_wmma_f32_16x16x32_bf16(false, a1, false, b0, (short)0, acc10, false, false);
    }
  }

  int col  = lane & 15;
  int row0 = (lane >> 4) << 3;          // C/D layout: VGPR v -> row v (+8 for upper lanes)
  float bv0 = bias ? bias[nt * 16 + col] : 0.f;
  float bv1 = (bias && hasN1) ? bias[(nt + 1) * 16 + col] : 0.f;
#pragma unroll
  for (int v = 0; v < 8; v++) {
    size_t r0 = (size_t)(mt * 16 + row0 + v) * N;
    size_t r1 = (size_t)((mt + 1) * 16 + row0 + v) * N;
    C[r0 + nt * 16 + col] = acc00[v] + bv0;
    C[r1 + nt * 16 + col] = acc10[v] + bv0;
    if (hasN1) {
      C[r0 + (nt + 1) * 16 + col] = acc01[v] + bv1;
      C[r1 + (nt + 1) * 16 + col] = acc11[v] + bv1;
    }
  }
}

// ---------------- depthwise 3x3 conv + SiLU; emit row- and col-major (f32 + bf16) ----------------
__global__ void k_conv_silu(const float* __restrict__ xz, const float* __restrict__ cw,
                            const float* __restrict__ cb,
                            float* __restrict__ xrow, float* __restrict__ xcol,
                            __bf16* __restrict__ xrowb, __bf16* __restrict__ xcolb) {
  size_t i = (size_t)blockIdx.x * blockDim.x + threadIdx.x;
  if (i >= (size_t)BB * LL * DI) return;
  int d = (int)(i % DI);
  int t = (int)(i / DI);
  int w = t % WW, h = (t / WW) % HH, b = t / (WW * HH);
  float acc = cb[d];
#pragma unroll
  for (int dy = -1; dy <= 1; dy++)
#pragma unroll
    for (int dx = -1; dx <= 1; dx++) {
      int hh = h + dy, ww = w + dx;
      if (hh >= 0 && hh < HH && ww >= 0 && ww < WW)
        acc += cw[d * 9 + (dy + 1) * 3 + (dx + 1)] *
               xz[((size_t)(b * HH + hh) * WW + ww) * (2 * DI) + d];
    }
  float s = acc / (1.f + __expf(-acc));         // SiLU
  size_t lr = (size_t)b * LL + h * WW + w;      // row-major l
  size_t lc = (size_t)b * LL + w * HH + h;      // col-major l
  xrow[lr * DI + d] = s;  xrowb[lr * DI + d] = f2bf(s);
  xcol[lc * DI + d] = s;  xcolb[lc * DI + d] = f2bf(s);
}

// ---------------- dt projection: (k,b,l,12) -> (k,b,l,384), K=12 plain FMA ----------------
__global__ void k_dtproj(const float* __restrict__ xd, const float* __restrict__ dtw,
                         float* __restrict__ dts) {
  size_t i = (size_t)blockIdx.x * blockDim.x + threadIdx.x;
  if (i >= (size_t)KD4 * BB * LL * DI) return;
  int d = (int)(i % DI);
  size_t t = i / DI;                      // (k*BB + b)*LL + l
  int k = (int)(t / ((size_t)BB * LL));
  const float* r = xd + t * CPAD;
  const float* w = dtw + ((size_t)k * DI + d) * DTR;
  float acc = 0.f;
#pragma unroll
  for (int j = 0; j < DTR; j++) acc += r[j] * w[j];
  dts[i] = acc;
}

// ================= chunked selective scan =================
// h_t = exp(delta_t*A) ⊙ h_{t-1} + delta_t*u_t*B_t ; y_t = C_t·h_t + D*u_t
// Chunk decay product: Π_t exp(delta_t*A_n) = exp(A_n * Σ delta_t) — one scalar/channel.
// Phase A: per-chunk local scan (h_in = 0), emit y_local, chunk-final h, Σdelta.
// Phase B: chain chunk states sequentially (32 steps), emit per-chunk carry-in.
// Phase C: replay carry term only (h ← a⊙h, y += C·h), atomically accumulate.

__global__ void __launch_bounds__(DI) k_scan_chunk(
    const float* __restrict__ xrow, const float* __restrict__ xcol,
    const float* __restrict__ dts,  const float* __restrict__ xd,
    const float* __restrict__ A_logs, const float* __restrict__ Ds,
    const float* __restrict__ dtb,
    float* __restrict__ yrow, float* __restrict__ ycol,
    float* __restrict__ hend, float* __restrict__ sumd) {
  int bc = blockIdx.x;                  // bk*NCH + c
  int bk = bc / NCH, c = bc % NCH;
  int k = bk / BB, b = bk % BB;
  int d = threadIdx.x;
  int kd = k * DI + d;
  float Arow[NS], h[NS];
#pragma unroll
  for (int n = 0; n < NS; n++) { Arow[n] = -__expf(A_logs[(size_t)kd * NS + n]); h[n] = 0.f; }
  float Dv = Ds[kd];
  float bias = dtb[kd];
  const float* u_base   = ((k & 1) ? xcol : xrow) + (size_t)b * LL * DI;
  const float* dts_base = dts + (size_t)bk * LL * DI;
  const float* xd_base  = xd  + (size_t)bk * LL * CPAD;
  float* y_base = ((k & 1) ? ycol : yrow) + (size_t)b * LL * DI;
  const bool rev = (k >= 2);
  float sdelta = 0.f;
  for (int t = c * CS; t < (c + 1) * CS; t++) {
    int l = rev ? (LL - 1 - t) : t;
    float u   = u_base[(size_t)l * DI + d];
    float dtv = dts_base[(size_t)l * DI + d] + bias;
    float delta = (dtv > 20.f) ? dtv : __logf(1.f + __expf(dtv));   // softplus
    sdelta += delta;
    const float* Bp = xd_base + (size_t)l * CPAD + DTR;
    const float* Cp = Bp + NS;
    float du = delta * u;
    float y = 0.f;
#pragma unroll
    for (int n = 0; n < NS; n++) {
      float dA = __expf(delta * Arow[n]);
      h[n] = h[n] * dA + du * Bp[n];
      y += h[n] * Cp[n];
    }
    y += Dv * u;
    atomicAdd(&y_base[(size_t)l * DI + d], y);
    int l2 = rev ? (l - 2) : (l + 2);
    if (l2 >= 0 && l2 < LL) {
      __builtin_prefetch(u_base + (size_t)l2 * DI + d, 0, 1);
      __builtin_prefetch(dts_base + (size_t)l2 * DI + d, 0, 1);
    }
  }
  float* he = hend + ((size_t)bc * DI + d) * NS;
#pragma unroll
  for (int n = 0; n < NS; n++) he[n] = h[n];
  sumd[(size_t)bc * DI + d] = sdelta;
}

__global__ void __launch_bounds__(DI) k_scan_carry(
    const float* __restrict__ hend, const float* __restrict__ sumd,
    const float* __restrict__ A_logs, float* __restrict__ hcarry) {
  int bk = blockIdx.x;                  // 32 blocks
  int k = bk / BB;
  int d = threadIdx.x;
  int kd = k * DI + d;
  float Arow[NS], H[NS];
#pragma unroll
  for (int n = 0; n < NS; n++) { Arow[n] = -__expf(A_logs[(size_t)kd * NS + n]); H[n] = 0.f; }
  for (int c = 0; c < NCH; c++) {
    size_t base = ((size_t)(bk * NCH + c) * DI + d) * NS;
#pragma unroll
    for (int n = 0; n < NS; n++) hcarry[base + n] = H[n];     // carry-in for chunk c
    float sd = sumd[(size_t)(bk * NCH + c) * DI + d];
    const float* he = hend + base;
#pragma unroll
    for (int n = 0; n < NS; n++)
      H[n] = __expf(Arow[n] * sd) * H[n] + he[n];             // chain chunk finals
  }
}

__global__ void __launch_bounds__(DI) k_scan_fix(
    const float* __restrict__ dts, const float* __restrict__ xd,
    const float* __restrict__ A_logs, const float* __restrict__ dtb,
    const float* __restrict__ hcarry,
    float* __restrict__ yrow, float* __restrict__ ycol) {
  int bc = blockIdx.x;
  int bk = bc / NCH, c = bc % NCH;
  if (c == 0) return;                   // zero carry into first chunk
  int k = bk / BB, b = bk % BB;
  int d = threadIdx.x;
  int kd = k * DI + d;
  float Arow[NS], h[NS];
  const float* hc = hcarry + ((size_t)bc * DI + d) * NS;
#pragma unroll
  for (int n = 0; n < NS; n++) { Arow[n] = -__expf(A_logs[(size_t)kd * NS + n]); h[n] = hc[n]; }
  float bias = dtb[kd];
  const float* dts_base = dts + (size_t)bk * LL * DI;
  const float* xd_base  = xd  + (size_t)bk * LL * CPAD;
  float* y_base = ((k & 1) ? ycol : yrow) + (size_t)b * LL * DI;
  const bool rev = (k >= 2);
  for (int t = c * CS; t < (c + 1) * CS; t++) {
    int l = rev ? (LL - 1 - t) : t;
    float dtv = dts_base[(size_t)l * DI + d] + bias;
    float delta = (dtv > 20.f) ? dtv : __logf(1.f + __expf(dtv));
    const float* Cp = xd_base + (size_t)l * CPAD + DTR + NS;
    float y = 0.f;
#pragma unroll
    for (int n = 0; n < NS; n++) {
      h[n] *= __expf(delta * Arow[n]);                        // carry term only
      y += h[n] * Cp[n];
    }
    atomicAdd(&y_base[(size_t)l * DI + d], y);
    int l2 = rev ? (l - 2) : (l + 2);
    if (l2 >= 0 && l2 < LL)
      __builtin_prefetch(dts_base + (size_t)l2 * DI + d, 0, 1);
  }
}

// ---------------- merge directions + LayerNorm + SiLU gate -> bf16 ----------------
__global__ void __launch_bounds__(DI) k_ln_gate(
    const float* __restrict__ yrow, const float* __restrict__ ycol,
    const float* __restrict__ xz, const float* __restrict__ lnw,
    const float* __restrict__ lnb, __bf16* __restrict__ yg) {
  int bl = blockIdx.x;                  // b*LL + l_row
  int b = bl / LL, l = bl % LL;
  int hh = l / WW, ww = l % WW;
  int lc = ww * HH + hh;                // col-major index of same pixel
  int d = threadIdx.x;
  float y = yrow[((size_t)b * LL + l) * DI + d] + ycol[((size_t)b * LL + lc) * DI + d];
  __shared__ float s1[DI], s2[DI];
  s1[d] = y; s2[d] = y * y;
  __syncthreads();
  if (d < 128) { s1[d] += s1[d + 128] + s1[d + 256]; s2[d] += s2[d + 128] + s2[d + 256]; }
  __syncthreads();
  for (int s = 64; s > 0; s >>= 1) {
    if (d < s) { s1[d] += s1[d + s]; s2[d] += s2[d + s]; }
    __syncthreads();
  }
  float mu   = s1[0] * (1.f / DI);
  float var  = s2[0] * (1.f / DI) - mu * mu;
  float rstd = rsqrtf(var + 1e-5f);
  float z  = xz[(size_t)bl * (2 * DI) + DI + d];
  float sz = z / (1.f + __expf(-z));
  float o  = ((y - mu) * rstd * lnw[d] + lnb[d]) * sz;
  yg[(size_t)bl * DI + d] = f2bf(o);
}

// ---------------- launch ----------------
extern "C" void kernel_launch(void* const* d_in, const int* in_sizes, int n_in,
                              void* d_out, int out_size, void* d_ws, size_t ws_size,
                              hipStream_t stream) {
  (void)in_sizes; (void)n_in; (void)out_size; (void)ws_size;
  const float* x      = (const float*)d_in[0];
  const float* inpw   = (const float*)d_in[1];
  const float* inpb   = (const float*)d_in[2];
  const float* convw  = (const float*)d_in[3];
  const float* convb  = (const float*)d_in[4];
  const float* xprojw = (const float*)d_in[5];
  const float* dtw    = (const float*)d_in[6];
  const float* dtb    = (const float*)d_in[7];
  const float* A_logs = (const float*)d_in[8];
  const float* Ds     = (const float*)d_in[9];
  const float* lnw    = (const float*)d_in[10];
  const float* lnb    = (const float*)d_in[11];
  const float* outpw  = (const float*)d_in[12];
  const float* outpb  = (const float*)d_in[13];
  float* out = (float*)d_out;

  char* ws = (char*)d_ws;
  size_t off = 0;
  auto alloc = [&](size_t bytes) -> void* {
    void* p = ws + off;
    off = (off + bytes + 255) & ~(size_t)255;
    return p;
  };
  float*  xz     = (float*) alloc((size_t)BL * 2 * DI * 4);
  float*  xrow   = (float*) alloc((size_t)BL * DI * 4);
  float*  xcol   = (float*) alloc((size_t)BL * DI * 4);
  float*  xd     = (float*) alloc((size_t)KD4 * BL * CPAD * 4);
  float*  dts    = (float*) alloc((size_t)KD4 * BL * DI * 4);
  float*  yrow   = (float*) alloc((size_t)BL * DI * 4);
  float*  ycol   = (float*) alloc((size_t)BL * DI * 4);
  float*  hend   = (float*) alloc((size_t)KD4 * BB * NCH * DI * NS * 4);
  float*  hcar   = (float*) alloc((size_t)KD4 * BB * NCH * DI * NS * 4);
  float*  sumd   = (float*) alloc((size_t)KD4 * BB * NCH * DI * 4);
  __bf16* xbf    = (__bf16*)alloc((size_t)BL * DM * 2);
  __bf16* xrowb  = (__bf16*)alloc((size_t)BL * DI * 2);
  __bf16* xcolb  = (__bf16*)alloc((size_t)BL * DI * 2);
  __bf16* inpwb  = (__bf16*)alloc((size_t)2 * DI * DM * 2);
  __bf16* xpwb   = (__bf16*)alloc((size_t)KD4 * CPAD * DI * 2);
  __bf16* outpwb = (__bf16*)alloc((size_t)DM * DI * 2);
  __bf16* ygb    = (__bf16*)alloc((size_t)BL * DI * 2);

  auto cvt = [&](const float* s, __bf16* d, size_t n) {
    k_f32_to_bf16<<<(unsigned)((n + 255) / 256), 256, 0, stream>>>(s, d, n);
  };
  auto gemm = [&](const __bf16* A, const __bf16* Bm, const float* bias, float* C,
                  int M, int N, int K) {
    long n2t = ((N / 16) + 1) / 2;
    long waves = (long)(M / 32) * n2t;
    long thr = waves * 32;
    k_gemm_wmma<<<(unsigned)((thr + 255) / 256), 256, 0, stream>>>(A, Bm, bias, C, M, N, K);
  };

  // weights / input to bf16
  cvt(x,     xbf,    (size_t)BL * DM);
  cvt(inpw,  inpwb,  (size_t)2 * DI * DM);
  cvt(outpw, outpwb, (size_t)DM * DI);
  k_pad_xprojw<<<(KD4 * CPAD * DI + 255) / 256, 256, 0, stream>>>(xprojw, xpwb);

  // in_proj: xz = X @ W^T + b  (32768 x 768 x 192)
  gemm(xbf, inpwb, inpb, xz, BL, 2 * DI, DM);

  // depthwise conv + SiLU -> row/col layouts
  {
    size_t n = (size_t)BB * LL * DI;
    k_conv_silu<<<(unsigned)((n + 255) / 256), 256, 0, stream>>>(xz, convw, convb,
                                                                 xrow, xcol, xrowb, xcolb);
  }

  // x_proj per direction (flip folded into backward scan): 32768 x 48 x 384
  for (int k = 0; k < KD4; k++) {
    const __bf16* A = (k & 1) ? xcolb : xrowb;
    gemm(A, xpwb + (size_t)k * CPAD * DI, nullptr,
         xd + (size_t)k * BB * LL * CPAD, BL, CPAD, DI);
  }

  // dt projection (K=12, plain FMA)
  {
    size_t n = (size_t)KD4 * BB * LL * DI;
    k_dtproj<<<(unsigned)((n + 255) / 256), 256, 0, stream>>>(xd, dtw, dts);
  }

  // zero direction-merge accumulators (atomicAdd targets)
  {
    size_t n = (size_t)BL * DI;
    k_zero<<<(unsigned)((n + 255) / 256), 256, 0, stream>>>(yrow, n);
    k_zero<<<(unsigned)((n + 255) / 256), 256, 0, stream>>>(ycol, n);
  }

  // chunked selective scan: A (local), B (chain carries), C (apply carries)
  k_scan_chunk<<<KD4 * BB * NCH, DI, 0, stream>>>(xrow, xcol, dts, xd, A_logs, Ds, dtb,
                                                  yrow, ycol, hend, sumd);
  k_scan_carry<<<KD4 * BB, DI, 0, stream>>>(hend, sumd, A_logs, hcar);
  k_scan_fix<<<KD4 * BB * NCH, DI, 0, stream>>>(dts, xd, A_logs, dtb, hcar, yrow, ycol);

  // merge + LayerNorm + gate -> bf16
  k_ln_gate<<<BL, DI, 0, stream>>>(yrow, ycol, xz, lnw, lnb, ygb);

  // out_proj: out = Y @ W^T + b  (32768 x 192 x 384)
  gemm(ygb, outpwb, outpb, out, BL, DM, DI);
}
